// LSTM_model_24704651887261
// MI455X (gfx1250) — matrix operand order repaired and checked by reading, not compiled, not run on hardware
//
#include <hip/hip_runtime.h>
#include <hip/hip_bf16.h>

typedef __attribute__((ext_vector_type(16))) _Float16 v16h;
typedef __attribute__((ext_vector_type(8)))  _Float16 v8h;
typedef __attribute__((ext_vector_type(8)))  float    v8f;
typedef __attribute__((ext_vector_type(4)))  float    v4f;

union V16 { v16h v; v8h h[2]; };

#define BB 64
#define TT 512
#define HH 256
#define G4H 1024

// LDS layout inside dynamic shared block (320 KB total):
//   [0      , 64KB )  : h double buffer, f16, 2 x (64 x 256)
//   [64KB   , 320KB)  : xg slab double buffer, f16, 2 x (64 x 1024)
#define HBUF_BYTES   (2 * BB * HH * 2)            // 65536
#define SLAB_HALVES  (BB * G4H)                   // 65536 halves = 128KB
#define SLAB_BYTES   (SLAB_HALVES * 2)
#define SMEM_BYTES   (HBUF_BYTES + 2 * SLAB_BYTES)  // 327680 = 320KB

__device__ inline v8f wmma_f16(v16h a, v16h b, v8f c) {
  return __builtin_amdgcn_wmma_f32_16x16x32_f16(
      /*neg_a=*/false, a, /*neg_b=*/false, b,
      /*c_mod=*/(short)0, c, /*reuse_a=*/false, /*reuse_b=*/false);
}

// ---------------------------------------------------------------------------
// Pack weight matrix w (4H x K, f32 row-major) into WMMA B-fragment-major f16:
// out[((nt*KT + kt)*32 + lane)*16 + h] = w[n*K + k]
//   n = nt*16 + (lane&15),  k = kt*32 + (lane>>4)*8 + (h&7) + (h>>3)*16
// ---------------------------------------------------------------------------
__global__ void pack_b_frags(const float* __restrict__ w, _Float16* __restrict__ out, int K) {
  int idx = blockIdx.x * blockDim.x + threadIdx.x;
  int total = G4H * K;
  if (idx >= total) return;
  int KT   = K >> 5;
  int h    = idx & 15;
  int lane = (idx >> 4) & 31;
  int frag = idx >> 9;
  int kt   = frag % KT;
  int nt   = frag / KT;
  int n = nt * 16 + (lane & 15);
  int k = kt * 32 + ((lane >> 4) * 8) + (h & 7) + ((h >> 3) * 16);
  out[idx] = (_Float16)w[(size_t)n * K + k];
}

__global__ void bias_sum(const float* __restrict__ a, const float* __restrict__ b,
                         float* __restrict__ o) {
  int i = blockIdx.x * blockDim.x + threadIdx.x;
  if (i < G4H) o[i] = a[i] + b[i];
}

__global__ void zero_f32(float* __restrict__ p, int n) {
  int i = blockIdx.x * blockDim.x + threadIdx.x;
  if (i < n) p[i] = 0.f;
}

// ---------------------------------------------------------------------------
// xgT[t][b][n] (f16) = (opt-relu) A(f32, BT x K) @ Bpacked(f16, K x 1024) + bias
// Time-major f16 output so the scan can stream one contiguous 128KB slab/step.
// block = 256 threads = 8 waves; each wave: one M-tile(16 bt rows) x 64 cols.
// grid.x = (BT/16) * 16 / 8 = 4096
// ---------------------------------------------------------------------------
__global__ __launch_bounds__(256)
void gemm_xg(const float* __restrict__ A, const _Float16* __restrict__ Bp,
             const float* __restrict__ bias, _Float16* __restrict__ xgT,
             int K, int do_relu) {
  int wave = threadIdx.x >> 5;
  int lane = threadIdx.x & 31;
  int w  = blockIdx.x * 8 + wave;   // global wave id
  int mt = w >> 4;                   // M tile (16 rows of BT)
  int ng = w & 15;                   // group of 4 N-tiles
  int KT = K >> 5;

  int arow  = mt * 16 + (lane & 15);
  int kbase = (lane >> 4) * 8;

  v8f acc[4] = {v8f{}, v8f{}, v8f{}, v8f{}};

  for (int kt = 0; kt < KT; ++kt) {
    // A fragment: lane holds row `arow`, k = kt*32 + kbase + {0..7, 16..23}
    const float* ap = A + (size_t)arow * K + kt * 32 + kbase;
    float fa[16];
    *(v4f*)(fa + 0)  = *(const v4f*)(ap + 0);
    *(v4f*)(fa + 4)  = *(const v4f*)(ap + 4);
    *(v4f*)(fa + 8)  = *(const v4f*)(ap + 16);
    *(v4f*)(fa + 12) = *(const v4f*)(ap + 20);
    V16 a;
#pragma unroll
    for (int i = 0; i < 16; ++i) {
      float x = fa[i];
      if (do_relu) x = fmaxf(x, 0.f);
      a.v[i] = (_Float16)x;
    }
#pragma unroll
    for (int g = 0; g < 4; ++g) {
      int nt = ng * 4 + g;
      const _Float16* bp = Bp + ((size_t)(nt * KT + kt) * 32 + lane) * 16;
      V16 b;
      b.h[0] = *(const v8h*)(bp);
      b.h[1] = *(const v8h*)(bp + 8);
      acc[g] = wmma_f16(a.v, b.v, acc[g]);
    }
  }

  // store time-major f16 with bias
  // C/D layout: n = 16*nt + (lane&15), bt-row = 16*mt + 8*(lane>>4) + r
  int mrow = mt * 16 + (lane >> 4) * 8;
#pragma unroll
  for (int g = 0; g < 4; ++g) {
    int n = (ng * 4 + g) * 16 + (lane & 15);
    float bv = bias[n];
#pragma unroll
    for (int r = 0; r < 8; ++r) {
      int row = mrow + r;                 // bt index
      int b   = row >> 9;                 // / TT
      int t   = row & (TT - 1);           // % TT
      xgT[((size_t)t * BB + b) * G4H + n] = (_Float16)(acc[g][r] + bv);
    }
  }
}

// ---------------------------------------------------------------------------
// LSTM scan: single persistent workgroup (1024 thr = 32 waves), T=512 steps.
// wave = (mh<<4)|jt : jt = hidden-column tile (16 cols), mh = batch half.
// Wave owns gate tiles {jt, 16+jt, 32+jt, 48+jt} == i/f/g/o for SAME columns
// -> cell update is intra-wave. h double-buffered in LDS (f16), c in regs.
// xg slab for step t+1 prefetched LDS-async (global_load_async_to_lds_b128)
// while step t computes; each wave drains its ASYNCcnt before the step barrier.
// ---------------------------------------------------------------------------
__global__ __launch_bounds__(1024)
void lstm_scan(const _Float16* __restrict__ xgT, const _Float16* __restrict__ Wp,
               const float* __restrict__ h0, const float* __restrict__ c0,
               float* __restrict__ y, float* __restrict__ hT, float* __restrict__ cT) {
  extern __shared__ char smem[];
  _Float16* hbuf = (_Float16*)smem;                       // 2 x 64x256
  _Float16* xgs  = (_Float16*)(smem + HBUF_BYTES);        // 2 x 64x1024

  int tid  = threadIdx.x;
  int wave = tid >> 5;
  int lane = tid & 31;
  int jt = wave & 15;   // H column tile
  int mh = wave >> 4;   // batch half -> 32 rows

  // async prefetch of one 128KB slab: 1024 threads x 8 x b128 (128B/thread)
  auto prefetch_slab = [&](int t) {
    const char* g = (const char*)xgT + (size_t)t * SLAB_BYTES + (size_t)tid * 128;
    unsigned l = (unsigned)(unsigned long long)
                 (smem + HBUF_BYTES + (size_t)(t & 1) * SLAB_BYTES + (size_t)tid * 128);
#pragma unroll
    for (int i = 0; i < 8; ++i) {
      unsigned long long ga = (unsigned long long)(g + i * 16);
      unsigned la = l + i * 16;
      asm volatile("global_load_async_to_lds_b128 %0, %1, off"
                   :: "v"(la), "v"(ga) : "memory");
    }
  };

  prefetch_slab(0);

  // init h buffer 0 from h0 (f32 -> f16)
  for (int i = tid; i < BB * HH; i += 1024) hbuf[i] = (_Float16)h0[i];

  int jcol = jt * 16 + (lane & 15);
  float c[2][8];
#pragma unroll
  for (int mti = 0; mti < 2; ++mti)
#pragma unroll
    for (int r = 0; r < 8; ++r) {
      int m = 32 * mh + 16 * mti + 8 * (lane >> 4) + r;
      c[mti][r] = c0[m * HH + jcol];
    }

  asm volatile("s_wait_asynccnt 0" ::: "memory");
  __syncthreads();

  for (int t = 0; t < TT; ++t) {
    // kick off next step's xg slab while we compute this one
    if (t + 1 < TT) prefetch_slab(t + 1);

    const _Float16* hsrc = hbuf + (size_t)(t & 1) * (BB * HH);
    _Float16*       hdst = hbuf + (size_t)((t + 1) & 1) * (BB * HH);
    const _Float16* xs   = xgs + (size_t)(t & 1) * SLAB_HALVES;

    v8f acc[2][4];
#pragma unroll
    for (int i = 0; i < 2; ++i)
#pragma unroll
      for (int g = 0; g < 4; ++g) acc[i][g] = v8f{};

    for (int kt = 0; kt < 8; ++kt) {
      V16 a[2];
#pragma unroll
      for (int mti = 0; mti < 2; ++mti) {
        int arow = 32 * mh + 16 * mti + (lane & 15);
        const _Float16* ap = hsrc + arow * HH + kt * 32 + (lane >> 4) * 8;
        a[mti].h[0] = *(const v8h*)(ap);
        a[mti].h[1] = *(const v8h*)(ap + 16);
      }
#pragma unroll
      for (int g = 0; g < 4; ++g) {
        int nt = g * 16 + jt;
        const _Float16* bp = Wp + ((size_t)(nt * 8 + kt) * 32 + lane) * 16;
        V16 b;
        b.h[0] = *(const v8h*)(bp);
        b.h[1] = *(const v8h*)(bp + 8);
#pragma unroll
        for (int mti = 0; mti < 2; ++mti)
          acc[mti][g] = wmma_f16(a[mti].v, b.v, acc[mti][g]);
      }
    }

    // cell update (intra-wave: i/f/g/o for column jcol all in this wave);
    // xg pre-activations come from the LDS slab (ds loads, f16)
#pragma unroll
    for (int mti = 0; mti < 2; ++mti) {
#pragma unroll
      for (int r = 0; r < 8; ++r) {
        int m = 32 * mh + 16 * mti + 8 * (lane >> 4) + r;   // batch index
        const _Float16* xr = xs + (size_t)m * G4H + jcol;
        float gi = acc[mti][0][r] + (float)xr[0 * HH];
        float gf = acc[mti][1][r] + (float)xr[1 * HH];
        float gg = acc[mti][2][r] + (float)xr[2 * HH];
        float go = acc[mti][3][r] + (float)xr[3 * HH];
        gi = 1.f / (1.f + __expf(-gi));
        gf = 1.f / (1.f + __expf(-gf));
        gg = tanhf(gg);
        go = 1.f / (1.f + __expf(-go));
        float cn = gf * c[mti][r] + gi * gg;
        c[mti][r] = cn;
        float hn = go * tanhf(cn);
        hdst[m * HH + jcol] = (_Float16)hn;
        y[((size_t)m * TT + t) * HH + jcol] = hn;
      }
    }

    // own async prefetch must land before the barrier publishes the slab
    asm volatile("s_wait_asynccnt 0" ::: "memory");
    __syncthreads();
  }

  // final state: h in hbuf parity TT&1 == 0; c in regs
#pragma unroll
  for (int mti = 0; mti < 2; ++mti)
#pragma unroll
    for (int r = 0; r < 8; ++r) {
      int m = 32 * mh + 16 * mti + 8 * (lane >> 4) + r;
      hT[m * HH + jcol] = (float)hbuf[m * HH + jcol];
      cT[m * HH + jcol] = c[mti][r];
    }
}

// ---------------------------------------------------------------------------
// out[b] = dot(y3[b, T-1, :], lin_w) + lin_b       (O == 1)
// ---------------------------------------------------------------------------
__global__ __launch_bounds__(256)
void final_linear(const float* __restrict__ y3, const float* __restrict__ lw,
                  const float* __restrict__ lb, float* __restrict__ out) {
  __shared__ float red[256];
  int b = blockIdx.x;
  int t = threadIdx.x;
  float v = y3[((size_t)b * TT + (TT - 1)) * HH + t] * lw[t];
  red[t] = v;
  __syncthreads();
  for (int s = 128; s > 0; s >>= 1) {
    if (t < s) red[t] += red[t + s];
    __syncthreads();
  }
  if (t == 0) out[b] = red[0] + lb[0];
}

// ---------------------------------------------------------------------------
extern "C" void kernel_launch(void* const* d_in, const int* in_sizes, int n_in,
                              void* d_out, int out_size, void* d_ws, size_t ws_size,
                              hipStream_t stream) {
  const float* x = (const float*)d_in[0];
  const float* wih[4] = {(const float*)d_in[1], (const float*)d_in[5],
                         (const float*)d_in[9], (const float*)d_in[13]};
  const float* whh[4] = {(const float*)d_in[2], (const float*)d_in[6],
                         (const float*)d_in[10], (const float*)d_in[14]};
  const float* bih[4] = {(const float*)d_in[3], (const float*)d_in[7],
                         (const float*)d_in[11], (const float*)d_in[15]};
  const float* bhh[4] = {(const float*)d_in[4], (const float*)d_in[8],
                         (const float*)d_in[12], (const float*)d_in[16]};
  const float* lin_w = (const float*)d_in[17];
  const float* lin_b = (const float*)d_in[18];

  const int Kin[4] = {64, 256, 256, 256};   // wih K per layer (l0a,l1a,l0b,l1b)

  // workspace carve-up
  char* p = (char*)d_ws;
  auto alloc = [&](size_t bytes) -> void* {
    void* r = (void*)p;
    p += (bytes + 255) & ~(size_t)255;
    return r;
  };
  _Float16*  xgT   = (_Float16*)alloc((size_t)TT * BB * G4H * 2);  // 64 MB, time-major
  float*     yA    = (float*)alloc((size_t)BB * TT * HH * 4);      // 32 MB
  float*     yB    = (float*)alloc((size_t)BB * TT * HH * 4);      // 32 MB
  _Float16*  wihP[4];
  _Float16*  whhP[4];
  float*     biasP[4];
  for (int l = 0; l < 4; ++l) {
    wihP[l]  = (_Float16*)alloc((size_t)G4H * Kin[l] * 2);
    whhP[l]  = (_Float16*)alloc((size_t)G4H * HH * 2);
    biasP[l] = (float*)alloc(G4H * 4);
  }
  float* zeroHC = (float*)alloc((size_t)BB * HH * 4);
  float* hT0 = (float*)alloc((size_t)BB * HH * 4);
  float* cT0 = (float*)alloc((size_t)BB * HH * 4);
  float* hT1 = (float*)alloc((size_t)BB * HH * 4);
  float* cT1 = (float*)alloc((size_t)BB * HH * 4);

  // ---- prep: pack weights, fuse biases, zero initial state ----
  for (int l = 0; l < 4; ++l) {
    int nih = G4H * Kin[l];
    pack_b_frags<<<(nih + 255) / 256, 256, 0, stream>>>(wih[l], wihP[l], Kin[l]);
    int nhh = G4H * HH;
    pack_b_frags<<<(nhh + 255) / 256, 256, 0, stream>>>(whh[l], whhP[l], HH);
    bias_sum<<<(G4H + 255) / 256, 256, 0, stream>>>(bih[l], bhh[l], biasP[l]);
  }
  zero_f32<<<(BB * HH + 255) / 256, 256, 0, stream>>>(zeroHC, BB * HH);

  const int gemm_blocks = (BB * TT / 16) * 16 / 8;   // 4096

  // ---- layer l0a: input relu(x), zero state ----
  gemm_xg<<<gemm_blocks, 256, 0, stream>>>(x, wihP[0], biasP[0], xgT, 64, 1);
  lstm_scan<<<1, 1024, SMEM_BYTES, stream>>>(xgT, whhP[0], zeroHC, zeroHC, yA, hT0, cT0);

  // ---- layer l1a: input y0, zero state ----
  gemm_xg<<<gemm_blocks, 256, 0, stream>>>(yA, wihP[1], biasP[1], xgT, 256, 0);
  lstm_scan<<<1, 1024, SMEM_BYTES, stream>>>(xgT, whhP[1], zeroHC, zeroHC, yB, hT1, cT1);

  // ---- layer l0b: input relu(y1), state (hT0, cT0) ----
  gemm_xg<<<gemm_blocks, 256, 0, stream>>>(yB, wihP[2], biasP[2], xgT, 256, 1);
  lstm_scan<<<1, 1024, SMEM_BYTES, stream>>>(xgT, whhP[2], hT0, cT0, yA, hT0, cT0);

  // ---- layer l1b: input y2, state (hT1, cT1) ----
  gemm_xg<<<gemm_blocks, 256, 0, stream>>>(yA, wihP[3], biasP[3], xgT, 256, 0);
  lstm_scan<<<1, 1024, SMEM_BYTES, stream>>>(xgT, whhP[3], hT1, cT1, yB, hT1, cT1);

  // ---- final linear head ----
  final_linear<<<BB, 256, 0, stream>>>(yB, lin_w, lin_b, (float*)d_out);
}
